// FP4Linear_67242007986735
// MI455X (gfx1250) — compile-verified, matrix operand order checked by reading, not run on hardware
//
#include <hip/hip_runtime.h>
#include <hip/hip_bf16.h>

typedef __attribute__((ext_vector_type(16))) _Float16 v16h;
typedef __attribute__((ext_vector_type(8)))  float    v8f;
typedef __attribute__((ext_vector_type(4)))  int      v4i;

#define AS1 __attribute__((address_space(1)))
#define AS3 __attribute__((address_space(3)))

#define TOKENS  2048
#define IN_DIM  4096
#define OUT_DIM 12288
#define QBLOCK  128

#define BM 128
#define BN 128
#define BK 128
#define KTILES (IN_DIM / BK)
#define LDSS 136                  // padded half-stride (272B) to dodge bank conflicts
#define TILE_HALFS (128 * LDSS)   // halves per LDS tile

#if defined(__has_builtin)
#if __has_builtin(__builtin_amdgcn_global_load_async_to_lds_b128)
#define HAVE_ASYNC_LDS 1
#endif
#endif

__device__ __forceinline__ void wait_async0() {
#if defined(__has_builtin)
#if __has_builtin(__builtin_amdgcn_s_wait_asynccnt)
    __builtin_amdgcn_s_wait_asynccnt(0);
#else
    asm volatile("s_wait_asynccnt 0" ::: "memory");
#endif
#else
    asm volatile("s_wait_asynccnt 0" ::: "memory");
#endif
}

// ---------------- pre-pass: x f32 -> f16 (one-shot, bandwidth-bound) ----------
__global__ __launch_bounds__(256)
void xcvt_kernel(const float* __restrict__ x, _Float16* __restrict__ xh) {
    const size_t i = ((size_t)blockIdx.x * 256 + threadIdx.x) * 8;
    const float4* p = reinterpret_cast<const float4*>(x + i);
    const float4 a = p[0], b = p[1];
    union { _Float16 h[8]; uint4 u; } w;
    w.h[0] = (_Float16)a.x; w.h[1] = (_Float16)a.y;
    w.h[2] = (_Float16)a.z; w.h[3] = (_Float16)a.w;
    w.h[4] = (_Float16)b.x; w.h[5] = (_Float16)b.y;
    w.h[6] = (_Float16)b.z; w.h[7] = (_Float16)b.w;
    *reinterpret_cast<uint4*>(xh + i) = w.u;
}

// ---------------- tile helpers ----------------
__device__ __forceinline__ void stage_w(const int* __restrict__ packed,
                                        const float* __restrict__ scales,
                                        const float* __restrict__ zeros,
                                        int n0, int k0, int tid,
                                        int4* wreg, float* sreg, float* zreg) {
    const int bidx = k0 >> 7; // quant-block index along K (BK == QBLOCK)
    #pragma unroll
    for (int q = 0; q < 8; ++q) {
        const int idx = tid + q * 256;   // 0..2047
        const int n   = idx >> 4;        // 0..127
        const int c4  = idx & 15;        // int4 chunk within row
        const size_t pbase = (((size_t)(n0 + n) * IN_DIM + k0) >> 1);
        wreg[q] = reinterpret_cast<const int4*>(packed + pbase)[c4];
        sreg[q] = scales[(size_t)(n0 + n) * (IN_DIM / QBLOCK) + bidx];
        zreg[q] = zeros [(size_t)(n0 + n) * (IN_DIM / QBLOCK) + bidx];
    }
}

__device__ __forceinline__ void dequant_w(_Float16* __restrict__ Bs, int tid,
                                          const int4* wreg, const float* sreg,
                                          const float* zreg) {
    #pragma unroll
    for (int q = 0; q < 8; ++q) {
        const int idx = tid + q * 256;
        const int n   = idx >> 4;
        const int c4  = idx & 15;
        const float s = sreg[q], z = zreg[q];
        const int b0 = wreg[q].x, b1 = wreg[q].y, b2 = wreg[q].z, b3 = wreg[q].w;
        union { _Float16 h[8]; uint4 u; } w;
        // reference unpack order: element 2j = high nibble, 2j+1 = low nibble
        w.h[0] = (_Float16)fmaf((float)((b0 >> 4) & 15), s, z);
        w.h[1] = (_Float16)fmaf((float)( b0       & 15), s, z);
        w.h[2] = (_Float16)fmaf((float)((b1 >> 4) & 15), s, z);
        w.h[3] = (_Float16)fmaf((float)( b1       & 15), s, z);
        w.h[4] = (_Float16)fmaf((float)((b2 >> 4) & 15), s, z);
        w.h[5] = (_Float16)fmaf((float)( b2       & 15), s, z);
        w.h[6] = (_Float16)fmaf((float)((b3 >> 4) & 15), s, z);
        w.h[7] = (_Float16)fmaf((float)( b3       & 15), s, z);
        *reinterpret_cast<uint4*>(&Bs[n * LDSS + (c4 << 3)]) = w.u;
    }
}

// X tile from pre-converted f16: async global->LDS copy (ASYNCcnt path)
__device__ __forceinline__ void load_x_f16(_Float16* __restrict__ As,
                                           const _Float16* __restrict__ xh,
                                           int m0, int k0, int tid) {
    #pragma unroll
    for (int q = 0; q < 8; ++q) {
        const int idx = tid + q * 256;   // 0..2047 b128 chunks
        const int m   = idx >> 4;        // 0..127
        const int k8  = (idx & 15) << 3; // halves 0..120
        const _Float16* g = xh + (size_t)(m0 + m) * IN_DIM + k0 + k8;
        _Float16* l = &As[m * LDSS + k8];
#if defined(HAVE_ASYNC_LDS)
        __builtin_amdgcn_global_load_async_to_lds_b128(
            (AS1 v4i*)g, (AS3 v4i*)l, 0, 0);
#else
        *reinterpret_cast<uint4*>(l) = *reinterpret_cast<const uint4*>(g);
#endif
    }
}

// X tile from f32 source (fallback when d_ws is too small): convert inline
__device__ __forceinline__ void load_x_f32(_Float16* __restrict__ As,
                                           const float* __restrict__ x,
                                           int m0, int k0, int tid) {
    const float4* xg = reinterpret_cast<const float4*>(x);
    #pragma unroll
    for (int q = 0; q < 16; ++q) {
        const int f  = tid + q * 256;
        const int m  = f >> 5;
        const int k4 = (f & 31) << 2;
        const float4 v = xg[((size_t)(m0 + m) * IN_DIM + k0 + k4) >> 2];
        union { _Float16 h[4]; uint2 u; } pk;
        pk.h[0] = (_Float16)v.x; pk.h[1] = (_Float16)v.y;
        pk.h[2] = (_Float16)v.z; pk.h[3] = (_Float16)v.w;
        *reinterpret_cast<uint2*>(&As[m * LDSS + k4]) = pk.u;
    }
}

__device__ __forceinline__ void mma_tile(const _Float16* __restrict__ As,
                                         const _Float16* __restrict__ Bs,
                                         int wm, int wn, int lmod, int lhi,
                                         v8f c[2][4]) {
    #pragma unroll
    for (int kk = 0; kk < BK; kk += 32) {
        v16h a[2], b[4];
        #pragma unroll
        for (int i = 0; i < 2; ++i) {
            const int m = wm * 32 + i * 16 + lmod;
            const uint4* pa =
                reinterpret_cast<const uint4*>(&As[m * LDSS + kk + lhi * 16]);
            union { v16h v; uint4 u[2]; } t;
            t.u[0] = pa[0]; t.u[1] = pa[1];
            a[i] = t.v;
        }
        #pragma unroll
        for (int j = 0; j < 4; ++j) {
            const int n = wn * 64 + j * 16 + lmod;
            const uint4* pb =
                reinterpret_cast<const uint4*>(&Bs[n * LDSS + kk + lhi * 16]);
            union { v16h v; uint4 u[2]; } t;
            t.u[0] = pb[0]; t.u[1] = pb[1];
            b[j] = t.v;
        }
        #pragma unroll
        for (int i = 0; i < 2; ++i)
            #pragma unroll
            for (int j = 0; j < 4; ++j)
                c[i][j] = __builtin_amdgcn_wmma_f32_16x16x32_f16(
                    false, a[i], false, b[j], (short)0, c[i][j], false, false);
    }
}

// ---------------- main GEMM: double-buffered LDS, async X, pipelined W --------
template <bool USE_XH>
__global__ __launch_bounds__(256)
void fp4_linear_wmma_kernel(const float*    __restrict__ x,
                            const _Float16* __restrict__ xh,
                            const int*      __restrict__ packed,
                            const float*    __restrict__ scales,
                            const float*    __restrict__ zeros,
                            const float*    __restrict__ bias,
                            float*          __restrict__ y) {
    extern __shared__ __align__(16) _Float16 smem[];
    // LDS layout (computed per-use, no static pointer tables):
    //   [0          .. TILE)    As buffer 0
    //   [TILE       .. 2*TILE)  As buffer 1
    //   [2*TILE     .. 3*TILE)  Bs buffer 0
    //   [3*TILE     .. 4*TILE)  Bs buffer 1

    const int tid  = threadIdx.x;
    const int lane = tid & 31;
    const int lmod = lane & 15;
    const int lhi  = lane >> 4;
    const int wid  = tid >> 5;
    const int wm   = wid >> 1;   // 0..3 -> 32 rows of M
    const int wn   = wid & 1;    // 0..1 -> 64 cols of N

    const int n0 = blockIdx.x * BN;
    const int m0 = blockIdx.y * BM;

    v8f c[2][4];
    #pragma unroll
    for (int i = 0; i < 2; ++i)
        #pragma unroll
        for (int j = 0; j < 4; ++j)
            c[i][j] = (v8f){0.f,0.f,0.f,0.f,0.f,0.f,0.f,0.f};

    int4  wreg[8];
    float sreg[8], zreg[8];

    // ---- prologue: fill buffer 0 ----
    if constexpr (USE_XH) load_x_f16(smem, xh, m0, 0, tid);
    else                  load_x_f32(smem, x,  m0, 0, tid);
    stage_w(packed, scales, zeros, n0, 0, tid, wreg, sreg, zreg);
    dequant_w(smem + 2 * TILE_HALFS, tid, wreg, sreg, zreg);
    if constexpr (USE_XH) wait_async0();
    __syncthreads();

    for (int kt = 0; kt < KTILES; ++kt) {
        const int cur = kt & 1, nxt = cur ^ 1;
        const bool has_next = (kt + 1 < KTILES);

        _Float16* As_cur = smem + cur * TILE_HALFS;
        _Float16* As_nxt = smem + nxt * TILE_HALFS;
        _Float16* Bs_cur = smem + (2 + cur) * TILE_HALFS;
        _Float16* Bs_nxt = smem + (2 + nxt) * TILE_HALFS;

        if (has_next) {
            // async X copy + W register staging overlap the WMMA work below
            if constexpr (USE_XH) load_x_f16(As_nxt, xh, m0, (kt + 1) * BK, tid);
            else                  load_x_f32(As_nxt, x,  m0, (kt + 1) * BK, tid);
            stage_w(packed, scales, zeros, n0, (kt + 1) * BK, tid, wreg, sreg, zreg);
        }

        mma_tile(As_cur, Bs_cur, wm, wn, lmod, lhi, c);

        if (has_next) {
            dequant_w(Bs_nxt, tid, wreg, sreg, zreg);
            if constexpr (USE_XH) wait_async0();
        }
        __syncthreads();
    }

    // ---- epilogue: documented C layout (VGPR r -> M = r + 8*lhi, N = lane&15)
    #pragma unroll
    for (int j = 0; j < 4; ++j) {
        const int n = n0 + wn * 64 + j * 16 + lmod;
        const float bv = bias[n];
        #pragma unroll
        for (int i = 0; i < 2; ++i) {
            const int mbase = m0 + wm * 32 + i * 16 + lhi * 8;
            #pragma unroll
            for (int r = 0; r < 8; ++r) {
                y[(size_t)(mbase + r) * OUT_DIM + n] = c[i][j][r] + bv;
            }
        }
    }
}

extern "C" void kernel_launch(void* const* d_in, const int* in_sizes, int n_in,
                              void* d_out, int out_size, void* d_ws, size_t ws_size,
                              hipStream_t stream) {
    const float* x      = (const float*)d_in[0];
    const int*   packed = (const int*)  d_in[1];
    const float* scales = (const float*)d_in[2];
    const float* zeros  = (const float*)d_in[3];
    const float* bias   = (const float*)d_in[4];
    float* y = (float*)d_out;

    const dim3 grid(OUT_DIM / BN, TOKENS / BM); // 96 x 16
    const dim3 block(256);
    const size_t shbytes = (size_t)4 * TILE_HALFS * sizeof(_Float16); // 136 KB

    const size_t xh_bytes = (size_t)TOKENS * IN_DIM * sizeof(_Float16);
    if (ws_size >= xh_bytes) {
        _Float16* xh = (_Float16*)d_ws;
        xcvt_kernel<<<(TOKENS * IN_DIM) / (256 * 8), 256, 0, stream>>>(x, xh);
        fp4_linear_wmma_kernel<true><<<grid, block, shbytes, stream>>>(
            x, xh, packed, scales, zeros, bias, y);
    } else {
        fp4_linear_wmma_kernel<false><<<grid, block, shbytes, stream>>>(
            x, nullptr, packed, scales, zeros, bias, y);
    }
    (void)in_sizes; (void)n_in; (void)out_size;
}
